// TimeAwareTemporalAttention_91018946937477
// MI455X (gfx1250) — matrix operand order repaired and checked by reading, not compiled, not run on hardware
//
#include <hip/hip_runtime.h>

#define D_MODEL 512
#define NUM_HEADS 8
#define HEAD_DIM 64
#define BATCH 32
#define SEQ 1024
#define BL (BATCH * SEQ)          // 32768 rows
#define ALPHA 0.5f
#define BETA 0.5f

typedef __bf16 bf16_t;
typedef __attribute__((ext_vector_type(8)))  __bf16 v8bf;
typedef __attribute__((ext_vector_type(16))) __bf16 v16bf;
typedef __attribute__((ext_vector_type(8)))  float  v8f;
typedef __attribute__((ext_vector_type(4)))  unsigned int u32x4;
typedef __attribute__((ext_vector_type(8)))  int i32x8;
typedef __attribute__((ext_vector_type(4)))  int i32x4;

// LDS tile: 16 rows x 512 bf16, padded 16B per 1024B row -> 1040B row stride
#define LDS_ROW_ELEMS 520

// Build a 16-element A/B fragment from two 16-byte chunks.
static __device__ inline v16bf load_frag(const bf16_t* p0, const bf16_t* p1) {
  v8bf lo = *(const v8bf*)p0;
  v8bf hi = *(const v8bf*)p1;
  return __builtin_shufflevector(lo, hi, 0,1,2,3,4,5,6,7,8,9,10,11,12,13,14,15);
}

static __device__ inline v8f wmma_bf16(v16bf a, v16bf b, v8f c) {
  // (neg_a, A, neg_b, B, c_mod, C, reuse_a, reuse_b)
  return __builtin_amdgcn_wmma_f32_16x16x32_bf16(false, a, false, b, (short)0, c,
                                                 false, false);
}

#if defined(__has_builtin)
#if __has_builtin(__builtin_amdgcn_tensor_load_to_lds) && \
    __has_builtin(__builtin_amdgcn_s_wait_tensorcnt)
#define HAVE_TDM 1
#endif
#endif

static __device__ inline unsigned int lds_byte_offset(const void* p) {
  // ptrtoint of an addrspace(3) pointer yields the byte offset from LDS base
  return (unsigned int)(size_t)(__attribute__((address_space(3))) const void*)p;
}

// TDM: DMA a 16x512 bf16 tile (row stride 512 elems) from global into LDS,
// padding 4 DWORDs after every 256 DWORDs so LDS rows land 1040B apart.
static __device__ inline void tdm_load_tile_16x512(const bf16_t* gptr,
                                                   unsigned int lds_off) {
#ifdef HAVE_TDM
  unsigned long long ga = (unsigned long long)(size_t)gptr;
  u32x4 g0;
  g0[0] = 1u;                                       // count=1, user mode
  g0[1] = lds_off;                                  // lds_addr
  g0[2] = (unsigned int)(ga & 0xffffffffu);         // global_addr[31:0]
  g0[3] = (unsigned int)((ga >> 32) & 0x1ffffffu)   // global_addr[56:32]
          | (2u << 30);                             // type = 2 (image)
  i32x8 g1;
  g1[0] = (int)((1u << 16)      // data_size = 2 bytes
                | (1u << 20)    // pad_enable
                | (7u << 22)    // pad_interval: 256 DWORDs (1024 B)
                | (3u << 25));  // pad_amount: 4 DWORDs (16 B)
  g1[1] = (int)(512u << 16);    // tensor_dim0 = 512 (low 16 bits)
  g1[2] = (int)(16u << 16);     // tensor_dim0 hi=0 | tensor_dim1 = 16
  g1[3] = (int)(512u << 16);    // tensor_dim1 hi=0 | tile_dim0 = 512
  g1[4] = 16;                   // tile_dim1 = 16, tile_dim2 = 0
  g1[5] = 512;                  // tensor_dim0_stride[31:0] = 512 elems
  g1[6] = 0;                    // stride hi | tensor_dim1_stride lo
  g1[7] = 0;
  i32x4 z4 = {0, 0, 0, 0};
#if __clang_major__ >= 23
  i32x8 z8 = {0, 0, 0, 0, 0, 0, 0, 0};
  __builtin_amdgcn_tensor_load_to_lds(g0, g1, z4, z4, z8, 0);
#else
  __builtin_amdgcn_tensor_load_to_lds(g0, g1, z4, z4, 0);
#endif
#else
  (void)gptr; (void)lds_off;
#endif
}

// Stage a 16x512 bf16 tile into a padded LDS buffer (all threads participate).
static __device__ inline void stage_tile(const bf16_t* gsrc, bf16_t* lds_tile) {
#ifdef HAVE_TDM
  if ((threadIdx.x >> 5) == 0) {
    tdm_load_tile_16x512(gsrc, lds_byte_offset(lds_tile));
    __builtin_amdgcn_s_wait_tensorcnt(0);
  }
  __syncthreads();
#else
  // cooperative fallback copy, preserving the padded layout
  for (int e = threadIdx.x * 8; e < 16 * D_MODEL; e += blockDim.x * 8) {
    const int r = e >> 9, c = e & (D_MODEL - 1);
    *(v8bf*)(lds_tile + r * LDS_ROW_ELEMS + c) = *(const v8bf*)(gsrc + r * D_MODEL + c);
  }
  __syncthreads();
#endif
}

// ---------------------------------------------------------------- f32 -> bf16
__global__ void cvt_bf16_kernel(const float* __restrict__ src,
                                bf16_t* __restrict__ dst, int n) {
  int i = blockIdx.x * blockDim.x + threadIdx.x;
  int stride = gridDim.x * blockDim.x;
  for (; i < n; i += stride) dst[i] = (bf16_t)src[i];
}

// ------------------------------------------------------- fused QKV projection
// y = x @ W^T + b.  One wave computes a 16(M) x 64(N) tile; the 16x512 x-tile
// is staged once per block into LDS by the Tensor Data Mover.
// blockIdx.z selects Q / K / V.  Q,K stored [B,H,L,hd]; V stored [B,H,hd,L].
__global__ void qkv_proj_kernel(const bf16_t* __restrict__ xb,
                                const bf16_t* __restrict__ Wq,
                                const bf16_t* __restrict__ Wk,
                                const bf16_t* __restrict__ Wv,
                                const float* __restrict__ bq,
                                const float* __restrict__ bk,
                                const float* __restrict__ bv,
                                bf16_t* __restrict__ Qo,
                                bf16_t* __restrict__ Ko,
                                bf16_t* __restrict__ Vt) {
  __shared__ __align__(128) bf16_t xtile[16 * LDS_ROW_ELEMS];
  const int lane = threadIdx.x & 31;
  const int wave = threadIdx.x >> 5;
  const int half = lane >> 4;
  const int l15  = lane & 15;
  const int akb  = half * 16;   // A-fragment K base
  const int bkb  = half * 16;   // B-fragment K base
  const int r0   = blockIdx.x * 16;
  const int nb   = wave * 64;
  const int z    = blockIdx.z;

  const bf16_t* W    = (z == 0) ? Wq : (z == 1) ? Wk : Wv;
  const float*  bias = (z == 0) ? bq : (z == 1) ? bk : bv;

  stage_tile(xb + (size_t)r0 * D_MODEL, xtile);

  v8f acc[4] = {};
  const bf16_t* arow = xtile + l15 * LDS_ROW_ELEMS;
  const bf16_t* wrow = W + (size_t)(nb + l15) * D_MODEL + bkb;

  // fully-unrolled 2-stage pipeline: compile-time ping-pong, no copies
  v16bf bbuf[2][4];
#pragma unroll
  for (int t = 0; t < 4; ++t) {
    const bf16_t* br = wrow + (size_t)t * 16 * D_MODEL;
    bbuf[0][t] = load_frag(br, br + 8);
  }
#pragma unroll
  for (int kk = 0; kk < D_MODEL; kk += 32) {
    const int s = (kk >> 5) & 1;
    if (kk + 32 < D_MODEL) {
#pragma unroll
      for (int t = 0; t < 4; ++t) {
        const bf16_t* br = wrow + (size_t)t * 16 * D_MODEL + kk + 32;
        bbuf[s ^ 1][t] = load_frag(br, br + 8);
      }
    }
    v16bf a = load_frag(arow + kk + akb, arow + kk + akb + 8);
#pragma unroll
    for (int t = 0; t < 4; ++t) acc[t] = wmma_bf16(a, bbuf[s][t], acc[t]);
  }

#pragma unroll
  for (int t = 0; t < 4; ++t) {
    const int ncol = nb + t * 16 + l15;
    const float bv_ = bias[ncol];
    const int h = ncol >> 6, d = ncol & 63;
#pragma unroll
    for (int i = 0; i < 8; ++i) {
      const int row = r0 + i + half * 8;
      const int bb = row >> 10, ll = row & (SEQ - 1);
      const bf16_t v = (bf16_t)(acc[t][i] + bv_);
      if (z == 2)
        Vt[((size_t)(bb * NUM_HEADS + h) * HEAD_DIM + d) * SEQ + ll] = v;
      else if (z == 0)
        Qo[((size_t)(bb * NUM_HEADS + h) * SEQ + ll) * HEAD_DIM + d] = v;
      else
        Ko[((size_t)(bb * NUM_HEADS + h) * SEQ + ll) * HEAD_DIM + d] = v;
    }
  }
}

// --------------------------------------------------------- flash attention
// One wave per (b,h,16-query tile).  8 waves / block.
__global__ void attn_kernel(const bf16_t* __restrict__ Q,
                            const bf16_t* __restrict__ K,
                            const bf16_t* __restrict__ Vt,
                            const float* __restrict__ tvals,
                            bf16_t* __restrict__ Ob) {
  __shared__ bf16_t pbuf[8][16 * 40];      // per-wave 16x32 P tile, stride 40
  const int lane = threadIdx.x & 31;
  const int wave = threadIdx.x >> 5;
  const int half = lane >> 4;
  const int l15  = lane & 15;
  const int akb  = half * 16;
  const int bkb  = half * 16;
  const int b  = blockIdx.z;
  const int h  = blockIdx.y;
  const int bh = b * NUM_HEADS + h;
  const int q0 = (blockIdx.x * 8 + wave) * 16;

  // Q fragments (16 x 64 = 2 A-frags), resident for the whole kernel
  const bf16_t* qrow = Q + ((size_t)bh * SEQ + (q0 + l15)) * HEAD_DIM;
  const v16bf qa0 = load_frag(qrow + akb,      qrow + akb + 8);
  const v16bf qa1 = load_frag(qrow + 32 + akb, qrow + 32 + akb + 8);

  float tq[8];
#pragma unroll
  for (int i = 0; i < 8; ++i) tq[i] = tvals[b * SEQ + q0 + i + half * 8];

  float m[8], lsum[8];
#pragma unroll
  for (int i = 0; i < 8; ++i) { m[i] = -3.0e38f; lsum[i] = 0.f; }
  v8f acc[4] = {};
  bf16_t* myp = &pbuf[wave][0];

  for (int k0 = 0; k0 <= q0 + 15; k0 += 32) {
    float sv[2][8];
#pragma unroll
    for (int f = 0; f < 2; ++f) {
      const int key = k0 + f * 16 + l15;
      const bf16_t* krow = K + ((size_t)bh * SEQ + key) * HEAD_DIM;
      v16bf kb0 = load_frag(krow + bkb,      krow + bkb + 8);       // d 0..31
      v16bf kb1 = load_frag(krow + 32 + bkb, krow + 32 + bkb + 8);  // d 32..63
      v8f s = {};
      s = wmma_bf16(qa0, kb0, s);
      s = wmma_bf16(qa1, kb1, s);
      const float tk = tvals[b * SEQ + key];
#pragma unroll
      for (int i = 0; i < 8; ++i) {
        const int qidx = q0 + i + half * 8;
        const float dt = fabsf(tq[i] - tk);
        const float omega = 1.0f / (1.0f + __expf(ALPHA * dt - BETA));
        const float val = s[i] * 0.125f * omega;   // 1/sqrt(64) = 0.125
        sv[f][i] = (key > qidx) ? -3.0e38f : val;
      }
    }
    // online softmax: row i lives in element i across one 16-lane half
    float cf[8];
#pragma unroll
    for (int i = 0; i < 8; ++i) {
      float x = fmaxf(sv[0][i], sv[1][i]);
      x = fmaxf(x, __shfl_xor(x, 1));
      x = fmaxf(x, __shfl_xor(x, 2));
      x = fmaxf(x, __shfl_xor(x, 4));
      x = fmaxf(x, __shfl_xor(x, 8));
      const float nm = fmaxf(m[i], x);
      cf[i] = __expf(m[i] - nm);
      m[i] = nm;
      const float p0 = __expf(sv[0][i] - nm);
      const float p1 = __expf(sv[1][i] - nm);
      sv[0][i] = p0; sv[1][i] = p1;
      float rs = p0 + p1;
      rs += __shfl_xor(rs, 1);
      rs += __shfl_xor(rs, 2);
      rs += __shfl_xor(rs, 4);
      rs += __shfl_xor(rs, 8);
      lsum[i] = cf[i] * lsum[i] + rs;
    }
#pragma unroll
    for (int t = 0; t < 4; ++t)
#pragma unroll
      for (int i = 0; i < 8; ++i) acc[t][i] *= cf[i];

    // re-fragment P (C/D layout) -> A layout via wave-private LDS tile
#pragma unroll
    for (int f = 0; f < 2; ++f)
#pragma unroll
      for (int i = 0; i < 8; ++i)
        myp[(i + half * 8) * 40 + f * 16 + l15] = (bf16_t)sv[f][i];
    const bf16_t* prow = myp + l15 * 40;
    v16bf pa = load_frag(prow + akb, prow + akb + 8);

    // PV: B-frag from transposed V is contiguous per lane
#pragma unroll
    for (int t = 0; t < 4; ++t) {
      const bf16_t* vrow =
          Vt + ((size_t)bh * HEAD_DIM + t * 16 + l15) * SEQ + k0 + bkb;
      v16bf vb = load_frag(vrow, vrow + 8);
      acc[t] = wmma_bf16(pa, vb, acc[t]);
    }
  }

#pragma unroll
  for (int i = 0; i < 8; ++i) lsum[i] = 1.0f / lsum[i];
#pragma unroll
  for (int t = 0; t < 4; ++t) {
    const int col = h * HEAD_DIM + t * 16 + l15;
#pragma unroll
    for (int i = 0; i < 8; ++i) {
      const int row = q0 + i + half * 8;
      Ob[((size_t)b * SEQ + row) * D_MODEL + col] = (bf16_t)(acc[t][i] * lsum[i]);
    }
  }
}

// ------------------------------------------------------- output projection
__global__ void out_proj_kernel(const bf16_t* __restrict__ Ob,
                                const bf16_t* __restrict__ Wo,
                                const float* __restrict__ bo,
                                float* __restrict__ out) {
  __shared__ __align__(128) bf16_t otile[16 * LDS_ROW_ELEMS];
  const int lane = threadIdx.x & 31;
  const int wave = threadIdx.x >> 5;
  const int half = lane >> 4;
  const int l15  = lane & 15;
  const int akb  = half * 16;
  const int bkb  = half * 16;
  const int r0   = blockIdx.x * 16;
  const int nb   = wave * 64;

  stage_tile(Ob + (size_t)r0 * D_MODEL, otile);

  v8f acc[4] = {};
  const bf16_t* arow = otile + l15 * LDS_ROW_ELEMS;
  const bf16_t* wrow = Wo + (size_t)(nb + l15) * D_MODEL + bkb;

  v16bf bbuf[2][4];
#pragma unroll
  for (int t = 0; t < 4; ++t) {
    const bf16_t* br = wrow + (size_t)t * 16 * D_MODEL;
    bbuf[0][t] = load_frag(br, br + 8);
  }
#pragma unroll
  for (int kk = 0; kk < D_MODEL; kk += 32) {
    const int s = (kk >> 5) & 1;
    if (kk + 32 < D_MODEL) {
#pragma unroll
      for (int t = 0; t < 4; ++t) {
        const bf16_t* br = wrow + (size_t)t * 16 * D_MODEL + kk + 32;
        bbuf[s ^ 1][t] = load_frag(br, br + 8);
      }
    }
    v16bf a = load_frag(arow + kk + akb, arow + kk + akb + 8);
#pragma unroll
    for (int t = 0; t < 4; ++t) acc[t] = wmma_bf16(a, bbuf[s][t], acc[t]);
  }

#pragma unroll
  for (int t = 0; t < 4; ++t) {
    const int ncol = nb + t * 16 + l15;
    const float bv_ = bo[ncol];
#pragma unroll
    for (int i = 0; i < 8; ++i) {
      const int row = r0 + i + half * 8;
      out[(size_t)row * D_MODEL + ncol] = acc[t][i] + bv_;
    }
  }
}

extern "C" void kernel_launch(void* const* d_in, const int* in_sizes, int n_in,
                              void* d_out, int out_size, void* d_ws, size_t ws_size,
                              hipStream_t stream) {
  const float* x  = (const float*)d_in[0];
  const float* tv = (const float*)d_in[1];
  const float* Wq = (const float*)d_in[2];
  const float* bq = (const float*)d_in[3];
  const float* Wk = (const float*)d_in[4];
  const float* bk = (const float*)d_in[5];
  const float* Wv = (const float*)d_in[6];
  const float* bv = (const float*)d_in[7];
  const float* Wo = (const float*)d_in[8];
  const float* bo = (const float*)d_in[9];
  float* out = (float*)d_out;

  char* ws = (char*)d_ws;
  size_t off = 0;
  const size_t XB = (size_t)BL * D_MODEL * sizeof(bf16_t);        // 32 MB
  const size_t WB = (size_t)D_MODEL * D_MODEL * sizeof(bf16_t);   // 0.5 MB
  bf16_t* xb  = (bf16_t*)(ws + off); off += XB;
  bf16_t* Wqb = (bf16_t*)(ws + off); off += WB;
  bf16_t* Wkb = (bf16_t*)(ws + off); off += WB;
  bf16_t* Wvb = (bf16_t*)(ws + off); off += WB;
  bf16_t* Wob = (bf16_t*)(ws + off); off += WB;
  bf16_t* Qb  = (bf16_t*)(ws + off); off += XB;   // [B,H,L,hd]
  bf16_t* Kb  = (bf16_t*)(ws + off); off += XB;   // [B,H,L,hd]
  bf16_t* Vt  = (bf16_t*)(ws + off); off += XB;   // [B,H,hd,L]
  bf16_t* Obf = (bf16_t*)(ws + off); off += XB;   // [B,L,D]

  cvt_bf16_kernel<<<4096, 256, 0, stream>>>(x, xb, BL * D_MODEL);
  cvt_bf16_kernel<<<256, 256, 0, stream>>>(Wq, Wqb, D_MODEL * D_MODEL);
  cvt_bf16_kernel<<<256, 256, 0, stream>>>(Wk, Wkb, D_MODEL * D_MODEL);
  cvt_bf16_kernel<<<256, 256, 0, stream>>>(Wv, Wvb, D_MODEL * D_MODEL);
  cvt_bf16_kernel<<<256, 256, 0, stream>>>(Wo, Wob, D_MODEL * D_MODEL);

  qkv_proj_kernel<<<dim3(BL / 16, 1, 3), 256, 0, stream>>>(
      xb, Wqb, Wkb, Wvb, bq, bk, bv, Qb, Kb, Vt);

  attn_kernel<<<dim3(SEQ / 128, NUM_HEADS, BATCH), 256, 0, stream>>>(
      Qb, Kb, Vt, tv, Obf);

  out_proj_kernel<<<dim3(BL / 16, 1, 1), 256, 0, stream>>>(Obf, Wob, bo, out);
}
